// GroupedQueryAttention_46042049413596
// MI455X (gfx1250) — matrix operand order repaired and checked by reading, not compiled, not run on hardware
//
#include <hip/hip_runtime.h>
#include <math.h>

typedef _Float16 h16;
typedef __attribute__((ext_vector_type(8)))  _Float16 v8h;
typedef __attribute__((ext_vector_type(16))) _Float16 v16h;
typedef __attribute__((ext_vector_type(8)))  float    v8f;

typedef __attribute__((ext_vector_type(4))) unsigned int u32x4_t;
typedef __attribute__((ext_vector_type(8))) int          i32x8_t;
typedef __attribute__((ext_vector_type(4))) int          i32x4_t;

#define B_   2
#define S_   2048
#define DM   2048
#define HQ   16
#define HKV  4
#define DK   128

// ---------------------------------------------------------------------------
// WMMA helpers (CDNA5 v_wmma_f32_16x16x32_f16, wave32)
// ---------------------------------------------------------------------------
__device__ __forceinline__ v8f wmma32(v16h a, v16h b, v8f c) {
  return __builtin_amdgcn_wmma_f32_16x16x32_f16(false, a, false, b, (short)0, c,
                                                false, false);
}

__device__ __forceinline__ v8f zerov8f() {
  v8f z;
#pragma unroll
  for (int i = 0; i < 8; ++i) z[i] = 0.f;
  return z;
}

__device__ __forceinline__ uint32_t lds_addr_of(const void* p) {
  return (uint32_t)(uintptr_t)(const __attribute__((address_space(3))) void*)p;
}

// ---------------------------------------------------------------------------
// CDNA5 async tile staging (GVS form: SGPR base + 32-bit VGPR byte offset),
// 16B per lane, tracked by ASYNCcnt.
// ---------------------------------------------------------------------------
__device__ __forceinline__ void async_ld16_s(void* lds, const void* base,
                                             uint32_t byte_off) {
  uint32_t laddr = lds_addr_of(lds);
  asm volatile("global_load_async_to_lds_b128 %0, %1, %2"
               :
               : "v"(laddr), "v"(byte_off), "s"((uint64_t)(uintptr_t)base)
               : "memory");
}
__device__ __forceinline__ void wait_async0() {
  asm volatile("s_wait_asynccnt 0x0" ::: "memory");
}

// ---------------------------------------------------------------------------
// Tensor Data Mover: 2D tile (16-bit elements) global -> LDS.
// D# per CDNA5 ISA ch.8: group0 = {count|flags, lds_addr, global_addr, type},
// group1 = {mask/data_size, dims, tile dims, dim0 stride}, higher groups zero.
// 6-arg builtin form (this toolchain): (u32x4, i32x8, i32x4, i32x4, i32x8, i32).
// ---------------------------------------------------------------------------
__device__ __forceinline__ void tdm_load_2d(uint32_t lds_byte, const void* gptr,
                                            uint32_t tile_d0,   // contig elems
                                            uint32_t tile_d1,   // rows
                                            uint64_t stride_d0  // elems
) {
  const uint64_t ga = (uint64_t)(uintptr_t)gptr;
  u32x4_t g0;
  g0[0] = 1u;  // count=1 (valid user descriptor)
  g0[1] = lds_byte;
  g0[2] = (uint32_t)ga;
  g0[3] = (uint32_t)((ga >> 32) & 0x01FFFFFFu) | (2u << 30);  // type=2 "image"

  union { uint64_t q[4]; i32x8_t v; } g1;
  const uint64_t td0 = tile_d0, td1 = tile_d1;
  // [17:16]=data_size(1->2B) | [79:48]=tensor_dim0 | [111:80]=tensor_dim1 |
  // [127:112]=tile_dim0 | [143:128]=tile_dim1 | [207:160]=tensor_dim0_stride
  g1.q[0] = (1ull << 16) | (td0 << 48);
  g1.q[1] = (td0 >> 16) | (td1 << 16) | ((td0 & 0xFFFFull) << 48);
  g1.q[2] = (td1 & 0xFFFFull) | ((stride_d0 & 0xFFFFFFFFull) << 32);
  g1.q[3] = (stride_d0 >> 32) & 0xFFFFull;

  i32x4_t z4;
#pragma unroll
  for (int i = 0; i < 4; ++i) z4[i] = 0;
  i32x8_t z8;
#pragma unroll
  for (int i = 0; i < 8; ++i) z8[i] = 0;
  __builtin_amdgcn_tensor_load_to_lds(g0, g1.v, z4, z4, z8, 0);
}
__device__ __forceinline__ void wait_tensor0() {
  __builtin_amdgcn_s_wait_tensorcnt(0);
}

// A-matrix fragment (16x32, row-major source, stride in halfs).
__device__ __forceinline__ v16h ld_frag_a(const h16* base, int stride, int lane) {
  const int l  = lane & 15;
  const int ko = (lane & 16) ? 8 : 0;
  const h16* p = base + (size_t)l * stride + ko;
  v8h lo = *(const v8h*)(p);
  v8h hi = *(const v8h*)(p + 16);
  v16h f;
#pragma unroll
  for (int i = 0; i < 8; ++i) { f[i] = lo[i]; f[i + 8] = hi[i]; }
  return f;
}

// B-matrix fragment (32x16). Source stored column-major as [N][K].
__device__ __forceinline__ v16h ld_frag_b(const h16* base, int stride, int lane) {
  const int l  = lane & 15;
  const int ko = (lane & 16) ? 8 : 0;
  const h16* p = base + (size_t)l * stride + ko;
  v8h lo = *(const v8h*)(p);
  v8h hi = *(const v8h*)(p + 16);
  v16h f;
#pragma unroll
  for (int i = 0; i < 8; ++i) { f[i] = lo[i]; f[i + 8] = hi[i]; }
  return f;
}

// ---------------------------------------------------------------------------
// One-time fp32 -> fp16 conversion (plain and transposing variants).
// ---------------------------------------------------------------------------
__global__ __launch_bounds__(256) void f32_to_f16_kernel(
    const float* __restrict__ in, h16* __restrict__ out, int n) {
  int i = blockIdx.x * 256 + threadIdx.x;
  if (i < n) out[i] = (h16)in[i];
}

// in: [K][N] row-major fp32 -> out: [N][K] fp16
__global__ __launch_bounds__(256) void f32_to_f16_tr_kernel(
    const float* __restrict__ in, h16* __restrict__ out, int K, int N) {
  int i = blockIdx.x * 256 + threadIdx.x;
  if (i < K * N) {
    int k = i / N, n = i % N;
    out[(size_t)n * K + k] = (h16)in[i];
  }
}

// ---------------------------------------------------------------------------
// Kernel 1: fused QKV projection + RoPE + RMSNorm epilogue.
// grid.x = (B*S)/64 row blocks, grid.y = 24 head-columns (16 Q, 4 K, 4 V).
// Weights pre-transposed [N][K]; async-staged, double-buffered LDS tiles.
// V is written transposed ([d][s]) for the attention PV stage.
// ---------------------------------------------------------------------------
__global__ __launch_bounds__(256) void qkv_rope_rms_kernel(
    const h16* __restrict__ xh,
    const h16* __restrict__ WqT, const h16* __restrict__ WkT,
    const h16* __restrict__ WvT,
    const float* __restrict__ qg, const float* __restrict__ kg,
    h16* __restrict__ Qb, h16* __restrict__ Kb, h16* __restrict__ Vtb) {
  __shared__ __align__(16) h16 lx[2][64][32];
  __shared__ __align__(16) h16 lw[2][128][32];
  __shared__ float lacc[64][128];

  const int tid  = threadIdx.x, lane = tid & 31, wv = tid >> 5;
  const int mi   = wv & 3, ns = wv >> 2;
  const int mblk = blockIdx.x, hc = blockIdx.y;

  const h16* Wt; int col0, kind;
  if (hc < 16)      { Wt = WqT; col0 = hc * 128;        kind = 0; }
  else if (hc < 20) { Wt = WkT; col0 = (hc - 16) * 128; kind = 1; }
  else              { Wt = WvT; col0 = (hc - 20) * 128; kind = 2; }

  const int row0 = mblk * 64;
  const int xr = tid >> 2, xc = (tid & 3) * 8;

  auto stage = [&](int buf, int k0) {
    async_ld16_s(&lx[buf][xr][xc], xh,
                 (uint32_t)(((unsigned)(row0 + xr) * DM + k0 + xc) * 2u));
#pragma unroll
    for (int it = 0; it < 2; ++it) {
      int idx = tid + it * 256;
      int n = idx >> 2, kc = (idx & 3) * 8;
      async_ld16_s(&lw[buf][n][kc], Wt,
                   (uint32_t)(((unsigned)(col0 + n) * DM + k0 + kc) * 2u));
    }
  };

  v8f acc[4];
#pragma unroll
  for (int j = 0; j < 4; ++j) acc[j] = zerov8f();

  stage(0, 0);
  for (int it = 0; it < DM / 32; ++it) {
    wait_async0();
    __syncthreads();
    if (it + 1 < DM / 32) stage((it + 1) & 1, (it + 1) * 32);
    const int buf = it & 1;
    v16h a = ld_frag_a(&lx[buf][mi * 16][0], 32, lane);
#pragma unroll
    for (int j = 0; j < 4; ++j) {
      v16h b = ld_frag_b(&lw[buf][(ns * 4 + j) * 16][0], 32, lane);
      acc[j] = wmma32(a, b, acc[j]);
    }
  }

  {  // dump accumulators to LDS
    const int l = lane & 15, mb = mi * 16 + ((lane & 16) ? 8 : 0);
#pragma unroll
    for (int j = 0; j < 4; ++j) {
      const int n = (ns * 4 + j) * 16 + l;
#pragma unroll
      for (int r = 0; r < 8; ++r) lacc[mb + r][n] = acc[j][r];
    }
  }
  __syncthreads();

  if (tid < 64) {  // one thread per row: RoPE then RMSNorm (reference order)
    const int r = tid;
    const int grow = row0 + r;
    const int b = grow / S_, s = grow % S_;
    float* rowp = &lacc[r][0];
    if (kind < 2) {
      const float pos = (float)s;
      for (int c = 0; c < 64; ++c) {
        float invf = __expf((float)c * -0.14391156f);  // 10000^(-c/64)
        float sn, cs;
        __sincosf(pos * invf, &sn, &cs);
        float t1 = rowp[c], t2 = rowp[c + 64];
        rowp[c]      = t1 * cs - t2 * sn;
        rowp[c + 64] = t2 * cs + t1 * sn;
      }
      float ssum = 0.f;
      for (int c = 0; c < 128; ++c) ssum += rowp[c] * rowp[c];
      const float rms = rsqrtf(ssum * (1.f / 128.f) + 1e-6f);
      const float* gam = (kind == 0) ? qg : kg;
      h16* dst = (kind == 0)
                     ? Qb + ((size_t)(b * HQ + hc) * S_ + s) * DK
                     : Kb + ((size_t)(b * HKV + (hc - 16)) * S_ + s) * DK;
      for (int c = 0; c < 128; ++c) dst[c] = (h16)(rowp[c] * rms * gam[c]);
    } else {  // V: store transposed [b][kvh][d][s]
      h16* dst = Vtb + ((size_t)(b * HKV + (hc - 20)) * DK) * S_ + s;
      for (int c = 0; c < 128; ++c) dst[(size_t)c * S_] = (h16)rowp[c];
    }
  }
}

// ---------------------------------------------------------------------------
// Kernel 2: causal flash attention. grid = (S/64 q-blocks, 16 q-heads, B).
// K / V^T tiles staged by the Tensor Data Mover (one D# each, wave 0 issues,
// TENSORcnt completion); Q fragments resident in registers; online softmax
// with O held in WMMA accumulators. Blocks above the diagonal skipped.
// ---------------------------------------------------------------------------
__global__ __launch_bounds__(256) void attn_kernel(
    const h16* __restrict__ Qb, const h16* __restrict__ Kb,
    const h16* __restrict__ Vtb, h16* __restrict__ Oh) {
  __shared__ __align__(16) h16 lK[64][128];   // K block [kv_row][d]
  __shared__ __align__(16) h16 lVt[128][64];  // V^T block [d][kv_row]
  __shared__ float lS[64][64];
  __shared__ __align__(16) h16 lP[64][64];
  __shared__ float lscale[64];
  __shared__ float lsum[64];

  const int tid = threadIdx.x, lane = tid & 31, wv = tid >> 5;
  const int mi = wv & 3, ns = wv >> 2;
  const int qblk = blockIdx.x, qh = blockIdx.y, b = blockIdx.z;
  const int kvh = qh >> 2;  // GROUP_SIZE = 4
  const int s0 = qblk * 64;

  const h16* qbase  = Qb + ((size_t)(b * HQ + qh) * S_ + s0) * DK;
  const h16* kbase  = Kb + (size_t)(b * HKV + kvh) * S_ * DK;
  const h16* vtbase = Vtb + (size_t)(b * HKV + kvh) * DK * S_;

  v16h qf[4];
#pragma unroll
  for (int kc = 0; kc < 4; ++kc)
    qf[kc] = ld_frag_a(qbase + (size_t)(mi * 16) * DK + kc * 32, DK, lane);

  v8f oacc[4];
#pragma unroll
  for (int j = 0; j < 4; ++j) oacc[j] = zerov8f();
  float m_i = -1e30f, l_i = 0.f;

  for (int kb0 = 0; kb0 <= s0; kb0 += 64) {
    if (wv == 0) {  // TDM: one descriptor per tile; EXEC-independent DMA
      tdm_load_2d(lds_addr_of(&lK[0][0]), kbase + (size_t)kb0 * DK,
                  /*d0=*/DK, /*d1=*/64, /*stride=*/DK);
      tdm_load_2d(lds_addr_of(&lVt[0][0]), vtbase + kb0,
                  /*d0=*/64, /*d1=*/DK, /*stride=*/S_);
      wait_tensor0();
    }
    __syncthreads();

    // S = Q @ K^T
    v8f sacc[2];
#pragma unroll
    for (int j = 0; j < 2; ++j) sacc[j] = zerov8f();
#pragma unroll
    for (int j = 0; j < 2; ++j) {
      const int ni = ns * 2 + j;
#pragma unroll
      for (int kc = 0; kc < 4; ++kc) {
        v16h bfr = ld_frag_b(&lK[ni * 16][kc * 32], 128, lane);
        sacc[j] = wmma32(qf[kc], bfr, sacc[j]);
      }
    }
    {  // scaled scores to LDS
      const int l = lane & 15, mb = mi * 16 + ((lane & 16) ? 8 : 0);
      const float sc = 0.088388347648f;  // 1/sqrt(128)
#pragma unroll
      for (int j = 0; j < 2; ++j) {
        const int n = (ns * 2 + j) * 16 + l;
#pragma unroll
        for (int r = 0; r < 8; ++r) lS[mb + r][n] = sacc[j][r] * sc;
      }
    }
    __syncthreads();

    if (tid < 64) {  // online softmax, one thread per query row
      const int r = tid;
      const int cend = (kb0 == s0) ? (r + 1) : 64;  // causal
      float bm = -1e30f;
      for (int c = 0; c < cend; ++c) bm = fmaxf(bm, lS[r][c]);
      const float newm = fmaxf(m_i, bm);
      const float resc = __expf(m_i - newm);
      float ps = 0.f;
      for (int c = 0; c < 64; ++c) {
        float pv = (c < cend) ? __expf(lS[r][c] - newm) : 0.f;
        lP[r][c] = (h16)pv;
        ps += pv;
      }
      l_i = l_i * resc + ps;
      m_i = newm;
      lscale[r] = resc;
    }
    __syncthreads();

    {  // rescale running O
      const int mb = mi * 16 + ((lane & 16) ? 8 : 0);
#pragma unroll
      for (int r = 0; r < 8; ++r) {
        const float f = lscale[mb + r];
#pragma unroll
        for (int j = 0; j < 4; ++j) oacc[j][r] *= f;
      }
    }
    v16h pf[2];
#pragma unroll
    for (int kc = 0; kc < 2; ++kc)
      pf[kc] = ld_frag_a(&lP[mi * 16][kc * 32], 64, lane);
#pragma unroll
    for (int j = 0; j < 4; ++j) {  // O += P @ V
      const int nd = ns * 4 + j;
#pragma unroll
      for (int kc = 0; kc < 2; ++kc) {
        v16h vf = ld_frag_b(&lVt[nd * 16][kc * 32], 64, lane);
        oacc[j] = wmma32(pf[kc], vf, oacc[j]);
      }
    }
    __syncthreads();
  }

  if (tid < 64) lsum[tid] = l_i;
  __syncthreads();

  {  // normalize and store O as fp16 in [b, s, H*D] layout
    const int l = lane & 15, mb = mi * 16 + ((lane & 16) ? 8 : 0);
#pragma unroll
    for (int r = 0; r < 8; ++r) {
      const float inv = 1.f / lsum[mb + r];
      const int s = s0 + mb + r;
      h16* dst = Oh + ((size_t)(b * S_ + s) * DM) + qh * DK;
#pragma unroll
      for (int j = 0; j < 4; ++j)
        dst[(ns * 4 + j) * 16 + l] = (h16)(oacc[j][r] * inv);
    }
  }
}

// ---------------------------------------------------------------------------
// Kernel 3: output projection (Wfc pre-transposed), fp32 result.
// grid = ((B*S)/64, DM/128). Async-staged, double-buffered tiles.
// ---------------------------------------------------------------------------
__global__ __launch_bounds__(256) void out_proj_kernel(
    const h16* __restrict__ Ah, const h16* __restrict__ WT,
    float* __restrict__ out) {
  __shared__ __align__(16) h16 la[2][64][32];
  __shared__ __align__(16) h16 lw[2][128][32];
  const int tid = threadIdx.x, lane = tid & 31, wv = tid >> 5;
  const int mi = wv & 3, ns = wv >> 2;
  const int row0 = blockIdx.x * 64;
  const int col0 = blockIdx.y * 128;
  const int ar = tid >> 2, ac = (tid & 3) * 8;

  auto stage = [&](int buf, int k0) {
    async_ld16_s(&la[buf][ar][ac], Ah,
                 (uint32_t)(((unsigned)(row0 + ar) * DM + k0 + ac) * 2u));
#pragma unroll
    for (int it = 0; it < 2; ++it) {
      int idx = tid + it * 256;
      int n = idx >> 2, kc = (idx & 3) * 8;
      async_ld16_s(&lw[buf][n][kc], WT,
                   (uint32_t)(((unsigned)(col0 + n) * DM + k0 + kc) * 2u));
    }
  };

  v8f acc[4];
#pragma unroll
  for (int j = 0; j < 4; ++j) acc[j] = zerov8f();

  stage(0, 0);
  for (int it = 0; it < DM / 32; ++it) {
    wait_async0();
    __syncthreads();
    if (it + 1 < DM / 32) stage((it + 1) & 1, (it + 1) * 32);
    const int buf = it & 1;
    v16h a = ld_frag_a(&la[buf][mi * 16][0], 32, lane);
#pragma unroll
    for (int j = 0; j < 4; ++j) {
      v16h b = ld_frag_b(&lw[buf][(ns * 4 + j) * 16][0], 32, lane);
      acc[j] = wmma32(a, b, acc[j]);
    }
  }
  const int l = lane & 15, mb = mi * 16 + ((lane & 16) ? 8 : 0);
#pragma unroll
  for (int j = 0; j < 4; ++j) {
    const int n = col0 + (ns * 4 + j) * 16 + l;
#pragma unroll
    for (int r = 0; r < 8; ++r)
      out[(size_t)(row0 + mb + r) * DM + n] = acc[j][r];
  }
}

// ---------------------------------------------------------------------------
extern "C" void kernel_launch(void* const* d_in, const int* in_sizes, int n_in,
                              void* d_out, int out_size, void* d_ws,
                              size_t ws_size, hipStream_t stream) {
  (void)in_sizes; (void)n_in; (void)out_size; (void)ws_size;
  const float* x   = (const float*)d_in[0];
  // d_in[1] is the tril mask: causality is applied analytically instead.
  const float* Wq  = (const float*)d_in[2];
  const float* Wk  = (const float*)d_in[3];
  const float* Wv  = (const float*)d_in[4];
  const float* Wfc = (const float*)d_in[5];
  const float* qg  = (const float*)d_in[6];
  const float* kg  = (const float*)d_in[7];
  float* out = (float*)d_out;

  h16* p = (h16*)d_ws;
  h16* xh   = p; p += (size_t)B_ * S_ * DM;
  h16* WqT  = p; p += (size_t)DM * DM;
  h16* WkT  = p; p += (size_t)512 * DM;
  h16* WvT  = p; p += (size_t)512 * DM;
  h16* WfT  = p; p += (size_t)DM * DM;
  h16* Qb   = p; p += (size_t)B_ * HQ * S_ * DK;
  h16* Kb   = p; p += (size_t)B_ * HKV * S_ * DK;
  h16* Vtb  = p; p += (size_t)B_ * HKV * DK * S_;  // [b][h][d][s]
  h16* Oh   = p;

  {
    int n = B_ * S_ * DM;
    f32_to_f16_kernel<<<(n + 255) / 256, 256, 0, stream>>>(x, xh, n);
  }
  auto cvtT = [&](const float* src, h16* dst, int K, int N) {
    int n = K * N;
    f32_to_f16_tr_kernel<<<(n + 255) / 256, 256, 0, stream>>>(src, dst, K, N);
  };
  cvtT(Wq,  WqT, DM, DM);
  cvtT(Wk,  WkT, DM, 512);
  cvtT(Wv,  WvT, DM, 512);
  cvtT(Wfc, WfT, DM, DM);

  qkv_rope_rms_kernel<<<dim3(B_ * S_ / 64, 24), 256, 0, stream>>>(
      xh, WqT, WkT, WvT, qg, kg, Qb, Kb, Vtb);
  attn_kernel<<<dim3(S_ / 64, HQ, B_), 256, 0, stream>>>(Qb, Kb, Vtb, Oh);
  out_proj_kernel<<<dim3(B_ * S_ / 64, DM / 128), 256, 0, stream>>>(Oh, WfT, out);
}
